// ParallelNetworks_82523501625564
// MI455X (gfx1250) — matrix operand order repaired and checked by reading, not compiled, not run on hardware
//
#include <hip/hip_runtime.h>

// ---------------------------------------------------------------------------
// Fused 3-layer batched MLP for MI455X (gfx1250), wave32 + WMMA bf16.
//   M=64 models, B=4096 batch, IN=64, H=256, OUT=1
//   out = relu(relu(xs@W1+b1)@W2+b2)@W3+b3
// One block = one model x 128 batch rows. h1 in LDS (bf16, K-permuted so the
// WMMA-C register layout stores as contiguous b128s), h2 only in accumulators.
// ---------------------------------------------------------------------------

typedef __bf16 bf16;
typedef bf16  v16bf __attribute__((ext_vector_type(16)));
typedef bf16  v8bf  __attribute__((ext_vector_type(8)));
typedef bf16  v4bf  __attribute__((ext_vector_type(4)));
typedef bf16  v2bf  __attribute__((ext_vector_type(2)));
typedef float v8f   __attribute__((ext_vector_type(8)));
typedef float v4f   __attribute__((ext_vector_type(4)));

#define N_MODELS 64
#define IN_DIM   64
#define H_DIM    256
#define BATCH    4096
#define OUT_DIM  1
#define BT       128                 // batch rows per block
#define THREADS  256                 // 8 waves
#define TILES_PER_MODEL (BATCH / BT) // 32

// LDS layout (bytes)
#define OFF_XS   0                               // [BT][IN]  bf16 : 16384
#define OFF_H1   (OFF_XS  + BT * IN_DIM * 2)     // [BT][H]   bf16 : 65536
#define OFF_W1T  (OFF_H1  + BT * H_DIM * 2)      // [H][IN]   bf16 : 32768
#define OFF_W2T  (OFF_W1T + H_DIM * IN_DIM * 2)  // [H][H]    bf16 : 131072
#define OFF_W3   (OFF_W2T + H_DIM * H_DIM * 2)   // [H]       f32  : 1024
#define OFF_B1   (OFF_W3  + H_DIM * 4)
#define OFF_B2   (OFF_B1  + H_DIM * 4)
#define LDS_BYTES (OFF_B2 + H_DIM * 4)           // 248832 B (<320KB WGP LDS)

// A fragment: 16-bit A 16x32 (M x K). lane&15 = row; lanes 0-15 hold
// K = k0..k0+15 consecutively, lanes 16-31 hold K = k0+16..k0+31.
// Source is row-major [row][K] bf16 in LDS.
__device__ __forceinline__ v16bf lds_frag_a(const bf16* __restrict__ base,
                                            int stride, int r0, int k0, int lane) {
    const int r  = r0 + (lane & 15);
    const int kh = k0 + ((lane >> 4) << 4); // +0 or +16
    v8bf lo = *(const v8bf*)(base + r * stride + kh);
    v8bf hi = *(const v8bf*)(base + r * stride + kh + 8);
    v16bf out;
#pragma unroll
    for (int i = 0; i < 8; ++i) { out[i] = lo[i]; out[i + 8] = hi[i]; }
    return out;
}

// B fragment: 16-bit B 32x16 (K x N). lane&15 = column; lanes 0-15 hold
// K = k0..k0+15 consecutively, lanes 16-31 hold K = k0+16..k0+31.
// Source is transposed row-major [N][K] bf16 in LDS.
__device__ __forceinline__ v16bf lds_frag_b(const bf16* __restrict__ baseT,
                                            int stride, int n0, int k0, int lane) {
    const int n  = n0 + (lane & 15);
    const int kb = k0 + ((lane >> 4) << 4); // +0 or +16
    v8bf lo = *(const v8bf*)(baseT + n * stride + kb);
    v8bf hi = *(const v8bf*)(baseT + n * stride + kb + 8);
    v16bf out;
#pragma unroll
    for (int i = 0; i < 8; ++i) { out[i] = lo[i]; out[i + 8] = hi[i]; }
    return out;
}

__global__ __launch_bounds__(THREADS)
void mlp64_fused_kernel(const float* __restrict__ xs_g,
                        const float* __restrict__ w1_g,
                        const float* __restrict__ b1_g,
                        const float* __restrict__ w2_g,
                        const float* __restrict__ b2_g,
                        const float* __restrict__ w3_g,
                        const float* __restrict__ b3_g,
                        float* __restrict__ out_g) {
    extern __shared__ __align__(16) char smem[];
    bf16*  xs_s  = (bf16*)(smem + OFF_XS);
    bf16*  h1_s  = (bf16*)(smem + OFF_H1);
    bf16*  w1t_s = (bf16*)(smem + OFF_W1T);
    bf16*  w2t_s = (bf16*)(smem + OFF_W2T);
    float* w3_s  = (float*)(smem + OFF_W3);
    float* b1_s  = (float*)(smem + OFF_B1);
    float* b2_s  = (float*)(smem + OFF_B2);

    const int tid   = threadIdx.x;
    const int lane  = tid & 31;
    const int wave  = tid >> 5;
    const int m     = blockIdx.x >> 5;   // model index   (TILES_PER_MODEL == 32)
    const int tile  = blockIdx.x & 31;   // batch tile index
    const int brow0 = tile * BT;

    // ---- stage xs tile (streamed once -> nontemporal), f32 -> bf16 ----
    {
        const float* xsm = xs_g + (size_t)(m * BATCH + brow0) * IN_DIM;
#pragma unroll
        for (int it = 0; it < (BT * IN_DIM) / (THREADS * 4); ++it) { // 8
            const int lin = (it * THREADS + tid) * 4;
            v4f v = __builtin_nontemporal_load((const v4f*)(xsm + lin));
            v4bf b;
            b[0] = (bf16)v[0]; b[1] = (bf16)v[1];
            b[2] = (bf16)v[2]; b[3] = (bf16)v[3];
            *(v4bf*)(xs_s + lin) = b;
        }
    }
    // ---- stage W1 transposed: global [k][n] f32 -> LDS [n][k] bf16 ----
    // Each thread handles a 2(k) x 4(n) block -> packed b32 stores.
    {
        const float* w1m = w1_g + (size_t)m * IN_DIM * H_DIM;
#pragma unroll
        for (int it = 0; it < (IN_DIM * H_DIM) / (THREADS * 8); ++it) { // 8
            const int w  = it * THREADS + tid;
            const int n  = (w & 63) * 4;
            const int k  = (w >> 6) * 2;
            v4f a = *(const v4f*)(w1m + k * H_DIM + n);
            v4f b = *(const v4f*)(w1m + (k + 1) * H_DIM + n);
#pragma unroll
            for (int j = 0; j < 4; ++j) {
                v2bf p; p[0] = (bf16)a[j]; p[1] = (bf16)b[j];
                *(v2bf*)(w1t_s + (n + j) * IN_DIM + k) = p;
            }
        }
    }
    // ---- stage W2 transposed with permuted K ordering ----
    // Storage slot s holds true K-row k(s) = ((s&15)<<4) | (s>>4)  (involution).
    // This makes the layer-1 epilogue's C-register columns land contiguously.
    // Adjacent slots (s, s+1), s even, hold true rows (k0, k0+16).
    {
        const float* w2m = w2_g + (size_t)m * H_DIM * H_DIM;
#pragma unroll 4
        for (int it = 0; it < (H_DIM * H_DIM) / (THREADS * 8); ++it) { // 32
            const int w  = it * THREADS + tid;
            const int n  = (w & 63) * 4;
            const int s  = (w >> 6) * 2;              // even storage slot
            const int k0 = ((s & 15) << 4) | (s >> 4); // true K row for slot s
            v4f a = *(const v4f*)(w2m + k0 * H_DIM + n);
            v4f b = *(const v4f*)(w2m + (k0 + 16) * H_DIM + n);
#pragma unroll
            for (int j = 0; j < 4; ++j) {
                v2bf p; p[0] = (bf16)a[j]; p[1] = (bf16)b[j];
                *(v2bf*)(w2t_s + (n + j) * H_DIM + s) = p;
            }
        }
    }
    // ---- biases + W3 (kept f32 for the final dot) ----
    w3_s[tid] = w3_g[m * H_DIM * OUT_DIM + tid];
    b1_s[tid] = b1_g[m * H_DIM + tid];
    b2_s[tid] = b2_g[m * H_DIM + tid];
    __syncthreads();

    // Each wave owns 16 batch rows; no further cross-wave dependencies.
    const int rowbase = wave * 16;
    const int ncol  = lane & 15;        // C-matrix column for this lane
    const int rhalf = (lane >> 4) << 3; // row offset 0 or 8 in C layout

    v8f acc[16];

    // ======================= Layer 1: [16x64] @ [64x256] ====================
#pragma unroll
    for (int nt = 0; nt < 16; ++nt)
#pragma unroll
        for (int i = 0; i < 8; ++i) acc[nt][i] = 0.0f;

#pragma unroll
    for (int kt = 0; kt < IN_DIM / 32; ++kt) { // 2
        v16bf a = lds_frag_a(xs_s, IN_DIM, rowbase, kt * 32, lane);
        v16bf bc = lds_frag_b(w1t_s, IN_DIM, 0, kt * 32, lane);
#pragma unroll
        for (int nt = 0; nt < 16; ++nt) {
            v16bf bn = (nt < 15) ? lds_frag_b(w1t_s, IN_DIM, (nt + 1) * 16, kt * 32, lane)
                                 : bc;
            acc[nt] = __builtin_amdgcn_wmma_f32_16x16x32_bf16(
                false, a, false, bc, (short)0, acc[nt], false, false);
            bc = bn;
        }
    }
    // bias + relu -> h1, stored with permuted columns: true col (nt*16+ncol)
    // goes to slot (ncol*16+nt), so each lane writes 16 contiguous bf16 / row.
    {
        float bias[16];
#pragma unroll
        for (int nt = 0; nt < 16; ++nt) bias[nt] = b1_s[nt * 16 + ncol];
#pragma unroll
        for (int i = 0; i < 8; ++i) {
            v8bf lo, hi;
#pragma unroll
            for (int nt = 0; nt < 8; ++nt) {
                float v = acc[nt][i] + bias[nt];
                lo[nt] = (bf16)(v > 0.0f ? v : 0.0f);
            }
#pragma unroll
            for (int nt = 8; nt < 16; ++nt) {
                float v = acc[nt][i] + bias[nt];
                hi[nt - 8] = (bf16)(v > 0.0f ? v : 0.0f);
            }
            bf16* dst = h1_s + (rowbase + rhalf + i) * H_DIM + ncol * 16;
            *(v8bf*)dst       = lo;   // 16B-aligned packed stores
            *(v8bf*)(dst + 8) = hi;
        }
    }

    // ====================== Layer 2: [16x256] @ [256x256] ===================
    // h1 and W2T share the same permuted K ordering, so this is exact.
#pragma unroll
    for (int nt = 0; nt < 16; ++nt)
#pragma unroll
        for (int i = 0; i < 8; ++i) acc[nt][i] = 0.0f;

#pragma unroll 2
    for (int kt = 0; kt < H_DIM / 32; ++kt) { // 8
        v16bf a = lds_frag_a(h1_s, H_DIM, rowbase, kt * 32, lane);
        v16bf bc = lds_frag_b(w2t_s, H_DIM, 0, kt * 32, lane);
#pragma unroll
        for (int nt = 0; nt < 16; ++nt) {
            v16bf bn = (nt < 15) ? lds_frag_b(w2t_s, H_DIM, (nt + 1) * 16, kt * 32, lane)
                                 : bc;
            acc[nt] = __builtin_amdgcn_wmma_f32_16x16x32_bf16(
                false, a, false, bc, (short)0, acc[nt], false, false);
            bc = bn;
        }
    }

    // ==== Layer 3 (OUT=1): bias+relu on accumulators, dot with W3, ==========
    // ==== 16-lane shuffle reduction; h2 never leaves registers.    ==========
    float p[8];
#pragma unroll
    for (int i = 0; i < 8; ++i) p[i] = 0.0f;
#pragma unroll
    for (int nt = 0; nt < 16; ++nt) {
        const float w    = w3_s[nt * 16 + ncol];
        const float bias = b2_s[nt * 16 + ncol];
#pragma unroll
        for (int i = 0; i < 8; ++i) {
            float h = acc[nt][i] + bias;
            h = h > 0.0f ? h : 0.0f;
            p[i] += h * w;
        }
    }
#pragma unroll
    for (int i = 0; i < 8; ++i) {
        p[i] += __shfl_xor(p[i], 1);
        p[i] += __shfl_xor(p[i], 2);
        p[i] += __shfl_xor(p[i], 4);
        p[i] += __shfl_xor(p[i], 8);
    }
    if (ncol == 0) { // lanes 0 (rows 0-7) and 16 (rows 8-15)
        const float b3v = b3_g[m];
        float* outp = out_g + (size_t)(m * BATCH + brow0 + rowbase + rhalf);
#pragma unroll
        for (int i = 0; i < 8; ++i)
            __builtin_nontemporal_store(p[i] + b3v, outp + i);
    }
}

extern "C" void kernel_launch(void* const* d_in, const int* in_sizes, int n_in,
                              void* d_out, int out_size, void* d_ws, size_t ws_size,
                              hipStream_t stream) {
    (void)in_sizes; (void)n_in; (void)d_ws; (void)ws_size; (void)out_size;
    const float* xs = (const float*)d_in[0];
    const float* W1 = (const float*)d_in[1];
    const float* b1 = (const float*)d_in[2];
    const float* W2 = (const float*)d_in[3];
    const float* b2 = (const float*)d_in[4];
    const float* W3 = (const float*)d_in[5];
    const float* b3 = (const float*)d_in[6];
    float* out = (float*)d_out;

    dim3 grid(N_MODELS * TILES_PER_MODEL); // 2048 blocks
    dim3 block(THREADS);
    mlp64_fused_kernel<<<grid, block, LDS_BYTES, stream>>>(
        xs, W1, b1, W2, b2, W3, b3, out);
}
